// Attention_65695819760299
// MI455X (gfx1250) — compile-verified, matrix operand order checked by reading, not run on hardware
//
#include <hip/hip_runtime.h>
#include <hip/hip_bf16.h>

#define B_    4
#define NSEQ  2048
#define DIM   1024
#define HEADS 16
#define DHEAD 64
#define INNER 1024
#define EPS   1e-5f

typedef __attribute__((ext_vector_type(16))) _Float16 v16h;
typedef __attribute__((ext_vector_type(8)))  _Float16 v8h;
typedef __attribute__((ext_vector_type(8)))  float    v8f;
typedef __attribute__((ext_vector_type(4)))  int      v4i;

#define WMMA_F16(a, b, c) \
  __builtin_amdgcn_wmma_f32_16x16x32_f16(false, (a), false, (b), (short)0, (c), false, false)

#define AS1 __attribute__((address_space(1)))
#define AS3 __attribute__((address_space(3)))

#if __has_builtin(__builtin_amdgcn_global_load_async_to_lds_b128) && \
    __has_builtin(__builtin_amdgcn_s_wait_asynccnt)
#define HAVE_ASYNC 1
#else
#define HAVE_ASYNC 0
#endif

// Copy 16 bytes global -> LDS. Async (ASYNCcnt-tracked, no VGPR staging) when
// the gfx1250 async builtins are available; plain load/store otherwise.
__device__ __forceinline__ void async_copy16(_Float16* lds, const _Float16* g) {
#if HAVE_ASYNC
  __builtin_amdgcn_global_load_async_to_lds_b128((AS1 v4i*)g, (AS3 v4i*)lds, 0, 0);
#else
  *(v8h*)lds = *(const v8h*)g;
#endif
}

__device__ __forceinline__ void async_wait() {
#if HAVE_ASYNC
  __builtin_amdgcn_s_wait_asynccnt(0);
#endif
}

// Load a 16x32 f16 A-fragment (or B-fragment from an [N][K]-transposed tile).
// Layout per CDNA5 ISA 7.12.2: lanes 0-15 hold rows 0-15 with K {0..7,16..23},
// lanes 16-31 hold rows 0-15 with K {8..15,24..31}. Two 16B loads per lane.
__device__ __forceinline__ v16h load_frag(const _Float16* base, int row0, int stride, int koff) {
  int lane = threadIdx.x & 31;
  int r = row0 + (lane & 15);
  int c = koff + ((lane >> 4) << 3);
  const _Float16* p = base + r * stride + c;
  v8h lo = *(const v8h*)p;
  v8h hi = *(const v8h*)(p + 16);
  v16h out;
#pragma unroll
  for (int i = 0; i < 8; ++i) { out[i] = lo[i]; out[i + 8] = hi[i]; }
  return out;
}

// ---------------------------------------------------------------- weight convert+transpose
// in: f32 [K][Ng] row-major  ->  out: f16 [Ng][K] (i.e. W^T), so GEMM B tiles
// are contiguous along K and can be staged with async b128 copies.
__global__ void cvt_transpose(const float* __restrict__ in, _Float16* __restrict__ out,
                              int K, int Ng) {
  int idx = blockIdx.x * blockDim.x + threadIdx.x;
  if (idx >= K * Ng) return;
  int n = idx / K;
  int k = idx - n * K;
  out[(size_t)n * K + k] = (_Float16)in[(size_t)k * Ng + n];
}

// ---------------------------------------------------------------- row LN (width 1024) -> f16
__global__ void ln_rows_1024(const float* __restrict__ x, const float* __restrict__ g,
                             const float* __restrict__ bta, _Float16* __restrict__ xn) {
  int row = blockIdx.x;
  int tid = threadIdx.x;
  const float* xr = x + (size_t)row * DIM;
  float v[4];
  float s = 0.f, ss = 0.f;
#pragma unroll
  for (int i = 0; i < 4; ++i) {
    v[i] = xr[tid + i * 256];
    s += v[i];
    ss += v[i] * v[i];
  }
#pragma unroll
  for (int m = 16; m >= 1; m >>= 1) {
    s += __shfl_xor(s, m, 32);
    ss += __shfl_xor(ss, m, 32);
  }
  __shared__ float smem[16];
  int wave = tid >> 5, lane = tid & 31;
  if (lane == 0) { smem[wave] = s; smem[8 + wave] = ss; }
  __syncthreads();
  if (wave == 0) {
    float a = (lane < 8) ? smem[lane] : 0.f;
    float b = (lane < 8) ? smem[8 + lane] : 0.f;
#pragma unroll
    for (int m = 4; m >= 1; m >>= 1) {
      a += __shfl_xor(a, m, 32);
      b += __shfl_xor(b, m, 32);
    }
    if (lane == 0) { smem[0] = a; smem[8] = b; }
  }
  __syncthreads();
  float mu = smem[0] * (1.f / DIM);
  float var = smem[8] * (1.f / DIM) - mu * mu;
  float rs = rsqrtf(var + EPS);
  _Float16* xo = xn + (size_t)row * DIM;
#pragma unroll
  for (int i = 0; i < 4; ++i) {
    int c = tid + i * 256;
    xo[c] = (_Float16)((v[i] - mu) * rs * g[c] + bta[c]);
  }
}

// ---------------------------------------------------------------- head LN (width 64)
// do_v == 0: in-place LN on Kh [bh][n][64]
// do_v == 1: LN on Vraw [bh][n][64] -> Vt [bh][64][n] (transposed for PV WMMA)
__global__ void ln_head(_Float16* __restrict__ Kio, const _Float16* __restrict__ Vin,
                        _Float16* __restrict__ Vt, const float* __restrict__ g,
                        const float* __restrict__ bta, int do_v) {
  int wave = threadIdx.x >> 5, lane = threadIdx.x & 31;
  int row = blockIdx.x * 8 + wave;            // [0, BH*NSEQ)
  int bh = row / NSEQ, n = row - bh * NSEQ;
  const _Float16* src = do_v ? (Vin + (size_t)row * DHEAD) : (Kio + (size_t)row * DHEAD);
  int d0 = lane * 2, d1 = d0 + 1;
  float v0 = (float)src[d0], v1 = (float)src[d1];
  float s = v0 + v1, ss = v0 * v0 + v1 * v1;
#pragma unroll
  for (int m = 16; m >= 1; m >>= 1) {
    s += __shfl_xor(s, m, 32);
    ss += __shfl_xor(ss, m, 32);
  }
  float mu = s * (1.f / DHEAD);
  float var = ss * (1.f / DHEAD) - mu * mu;
  float rs = rsqrtf(var + EPS);
  float o0 = (v0 - mu) * rs * g[d0] + bta[d0];
  float o1 = (v1 - mu) * rs * g[d1] + bta[d1];
  if (do_v) {
    Vt[((size_t)(bh * DHEAD + d0)) * NSEQ + n] = (_Float16)o0;
    Vt[((size_t)(bh * DHEAD + d1)) * NSEQ + n] = (_Float16)o1;
  } else {
    Kio[(size_t)row * DHEAD + d0] = (_Float16)o0;
    Kio[(size_t)row * DHEAD + d1] = (_Float16)o1;
  }
}

// ---------------------------------------------------------------- WMMA GEMM
// C[M,Ng] = A[M,K] * Wt[Ng,K]^T.  Block tile 128(M) x 128(N), 8 waves
// (16 M-rows x 128 N each). K step 32; both tiles staged via async b128 to LDS.
// mode 0: outf = C + bias (fp32).  mode 1: scatter f16 into Qh/Kh/Vraw head layout.
__global__ void gemm_wmma(const _Float16* __restrict__ A, const _Float16* __restrict__ Wt,
                          int M, int Ng, int K, int mode, float* __restrict__ outf,
                          const float* __restrict__ bias, _Float16* __restrict__ Qh,
                          _Float16* __restrict__ Kh, _Float16* __restrict__ Vr) {
  __shared__ _Float16 ldsA[128 * 32];
  __shared__ _Float16 ldsB[128 * 32];
  int tid = threadIdx.x;
  int wave = tid >> 5;
  int lane = tid & 31;
  int n0 = blockIdx.x * 128;
  int m0 = blockIdx.y * 128;

  v8f acc[8] = {};

  int row = tid >> 1;              // 0..127
  int ch = (tid & 1) * 16;         // 0 or 16
  const _Float16* srcA = A  + (size_t)(m0 + row) * K + ch;
  const _Float16* srcB = Wt + (size_t)(n0 + row) * K + ch;

  for (int k0 = 0; k0 < K; k0 += 32) {
    async_copy16(ldsA + row * 32 + ch,     srcA + k0);
    async_copy16(ldsA + row * 32 + ch + 8, srcA + k0 + 8);
    async_copy16(ldsB + row * 32 + ch,     srcB + k0);
    async_copy16(ldsB + row * 32 + ch + 8, srcB + k0 + 8);
    if (k0 + 32 < K) {
      __builtin_prefetch(srcA + k0 + 32, 0, 3);
      __builtin_prefetch(srcB + k0 + 32, 0, 3);
    }
    async_wait();
    __syncthreads();
    v16h a = load_frag(ldsA, wave * 16, 32, 0);
#pragma unroll
    for (int t = 0; t < 8; ++t) {
      v16h b = load_frag(ldsB, t * 16, 32, 0);
      acc[t] = WMMA_F16(a, b, acc[t]);
    }
    __syncthreads();
  }

  int nlo = lane & 15;
  int mhi = (lane >> 4) * 8;
  if (mode == 0) {
#pragma unroll
    for (int t = 0; t < 8; ++t) {
      int n = n0 + t * 16 + nlo;
      float bv = bias[n];
#pragma unroll
      for (int r = 0; r < 8; ++r) {
        int m = m0 + wave * 16 + mhi + r;
        outf[(size_t)m * Ng + n] = acc[t][r] + bv;
      }
    }
  } else {
#pragma unroll
    for (int t = 0; t < 8; ++t) {
      int col = n0 + t * 16 + nlo;
      int which = col >> 10;
      int rem = col & 1023;
      int h = rem >> 6, d = rem & 63;
      _Float16* dst = (which == 0) ? Qh : (which == 1) ? Kh : Vr;
      float sc = (which == 0) ? 0.125f : 1.f;  // fold 1/sqrt(64) into Q
      size_t coloff = (size_t)h * NSEQ * DHEAD + d;
#pragma unroll
      for (int r = 0; r < 8; ++r) {
        int m = m0 + wave * 16 + mhi + r;
        int bb = m >> 11, nn = m & 2047;
        dst[(size_t)bb * HEADS * NSEQ * DHEAD + coloff + (size_t)nn * DHEAD] =
            (_Float16)(acc[t][r] * sc);
      }
    }
  }
}

// ---------------------------------------------------------------- flash attention
// One block = 128 q-rows of one (b,h); 8 waves x 16 rows. Online softmax,
// S = Q K^T and O += P V via WMMA f16->f32. Output [b][n][h*64+d] f16.
__global__ void attn_kernel(const _Float16* __restrict__ Qh, const _Float16* __restrict__ Kh,
                            const _Float16* __restrict__ Vt, _Float16* __restrict__ Ao) {
  __shared__ _Float16 ldsK[32 * 64];
  __shared__ _Float16 ldsV[64 * 32];
  __shared__ _Float16 ldsP[8][16 * 32];
  int tid = threadIdx.x, wave = tid >> 5, lane = tid & 31;
  int nlo = lane & 15, half8 = (lane >> 4) * 8;
  int bh = blockIdx.x >> 4;
  int q0 = (blockIdx.x & 15) * 128 + wave * 16;

  const _Float16* Qbase = Qh + ((size_t)bh * NSEQ + q0) * DHEAD;
  v16h aQ0 = load_frag(Qbase, 0, DHEAD, 0);
  v16h aQ1 = load_frag(Qbase, 0, DHEAD, 32);

  v8f o[4] = {};
  float mrow[8], lrow[8];
#pragma unroll
  for (int r = 0; r < 8; ++r) { mrow[r] = -3.0e38f; lrow[r] = 0.f; }

  int rk = tid >> 3, chk = (tid & 7) * 8;   // K tile 32x64: one b128 per thread
  int rd = tid >> 2, chv = (tid & 3) * 8;   // V^T tile 64x32: one b128 per thread
  const _Float16* srcK = Kh + ((size_t)bh * NSEQ + rk) * DHEAD + chk;
  const _Float16* srcV = Vt + ((size_t)bh * DHEAD + rd) * NSEQ + chv;

  for (int kb = 0; kb < NSEQ; kb += 32) {
    __syncthreads();
    async_copy16(ldsK + rk * 64 + chk, srcK + (size_t)kb * DHEAD);
    async_copy16(ldsV + rd * 32 + chv, srcV + kb);
    if (kb + 32 < NSEQ) {
      __builtin_prefetch(srcK + (size_t)(kb + 32) * DHEAD, 0, 3);
      __builtin_prefetch(srcV + kb + 32, 0, 3);
    }
    async_wait();
    __syncthreads();

    // S = Q K^T : two 16x16 f32 tiles (keys 0-15, 16-31), K-dim = d = 64
    v8f s0 = {}, s1 = {};
    {
      v16h b0 = load_frag(ldsK, 0, 64, 0);
      v16h b1 = load_frag(ldsK, 16, 64, 0);
      s0 = WMMA_F16(aQ0, b0, s0);
      s1 = WMMA_F16(aQ0, b1, s1);
      b0 = load_frag(ldsK, 0, 64, 32);
      b1 = load_frag(ldsK, 16, 64, 32);
      s0 = WMMA_F16(aQ1, b0, s0);
      s1 = WMMA_F16(aQ1, b1, s1);
    }

    // Online softmax in C-tile layout; row reductions stay inside 16-lane halves.
    _Float16* pw = ldsP[wave];
#pragma unroll
    for (int r = 0; r < 8; ++r) {
      float mx = fmaxf(s0[r], s1[r]);
#pragma unroll
      for (int m = 8; m >= 1; m >>= 1) mx = fmaxf(mx, __shfl_xor(mx, m, 32));
      float mn = fmaxf(mrow[r], mx);
      float alpha = __expf(mrow[r] - mn);
      float p0 = __expf(s0[r] - mn);
      float p1 = __expf(s1[r] - mn);
      float rsum = p0 + p1;
#pragma unroll
      for (int m = 8; m >= 1; m >>= 1) rsum += __shfl_xor(rsum, m, 32);
      lrow[r] = lrow[r] * alpha + rsum;
      mrow[r] = mn;
#pragma unroll
      for (int t = 0; t < 4; ++t) o[t][r] *= alpha;
      int prow = half8 + r;  // stage P (C layout -> row-major) for A-fragment reload
      pw[prow * 32 + nlo] = (_Float16)p0;
      pw[prow * 32 + 16 + nlo] = (_Float16)p1;
    }

    // O += P V : K-dim = 32 keys, 4 d-slices of 16
    v16h aP = load_frag(pw, 0, 32, 0);
#pragma unroll
    for (int t = 0; t < 4; ++t) {
      v16h bv = load_frag(ldsV, t * 16, 32, 0);
      o[t] = WMMA_F16(aP, bv, o[t]);
    }
  }

  int b = bh >> 4, h = bh & 15;
#pragma unroll
  for (int r = 0; r < 8; ++r) {
    float inv = 1.f / lrow[r];
    int q = q0 + half8 + r;
    size_t rowoff = ((size_t)b * NSEQ + q) * INNER + h * DHEAD;
#pragma unroll
    for (int t = 0; t < 4; ++t)
      Ao[rowoff + t * 16 + nlo] = (_Float16)(o[t][r] * inv);
  }
}

// ---------------------------------------------------------------- launch
extern "C" void kernel_launch(void* const* d_in, const int* in_sizes, int n_in,
                              void* d_out, int out_size, void* d_ws, size_t ws_size,
                              hipStream_t stream) {
  const float* x       = (const float*)d_in[0];
  const float* norm_g  = (const float*)d_in[1];
  const float* norm_b  = (const float*)d_in[2];
  const float* w_qkv   = (const float*)d_in[3];
  const float* normk_g = (const float*)d_in[4];
  const float* normk_b = (const float*)d_in[5];
  const float* normv_g = (const float*)d_in[6];
  const float* normv_b = (const float*)d_in[7];
  const float* w_out   = (const float*)d_in[8];
  const float* b_out   = (const float*)d_in[9];
  float* out = (float*)d_out;

  const size_t ROWS = (size_t)B_ * NSEQ;                    // 8192
  const size_t HEADSZ = (size_t)B_ * HEADS * NSEQ * DHEAD;  // 8M elements

  char* ws = (char*)d_ws;
  size_t off = 0;
  _Float16* xn16    = (_Float16*)(ws + off); off += ROWS * DIM * 2;
  _Float16* wqkv16t = (_Float16*)(ws + off); off += (size_t)DIM * 3 * INNER * 2;
  _Float16* wout16t = (_Float16*)(ws + off); off += (size_t)INNER * DIM * 2;
  _Float16* Qh      = (_Float16*)(ws + off); off += HEADSZ * 2;
  _Float16* Kh      = (_Float16*)(ws + off); off += HEADSZ * 2;
  _Float16* Vr      = (_Float16*)(ws + off); off += HEADSZ * 2;
  _Float16* Vt      = (_Float16*)(ws + off); off += HEADSZ * 2;
  _Float16* Ao      = (_Float16*)(ws + off); off += ROWS * INNER * 2;

  int nqkv = DIM * 3 * INNER;
  int nout = INNER * DIM;
  cvt_transpose<<<(nqkv + 255) / 256, 256, 0, stream>>>(w_qkv, wqkv16t, DIM, 3 * INNER);
  cvt_transpose<<<(nout + 255) / 256, 256, 0, stream>>>(w_out, wout16t, INNER, DIM);

  ln_rows_1024<<<(int)ROWS, 256, 0, stream>>>(x, norm_g, norm_b, xn16);

  gemm_wmma<<<dim3(3 * INNER / 128, ROWS / 128), 256, 0, stream>>>(
      xn16, wqkv16t, (int)ROWS, 3 * INNER, DIM, 1, nullptr, nullptr, Qh, Kh, Vr);

  int lnrows = B_ * HEADS * NSEQ;  // 131072 rows of 64
  ln_head<<<lnrows / 8, 256, 0, stream>>>(Kh, nullptr, nullptr, normk_g, normk_b, 0);
  ln_head<<<lnrows / 8, 256, 0, stream>>>(nullptr, Vr, Vt, normv_g, normv_b, 1);

  attn_kernel<<<B_ * HEADS * (NSEQ / 128), 256, 0, stream>>>(Qh, Kh, Vt, Ao);

  gemm_wmma<<<dim3(DIM / 128, ROWS / 128), 256, 0, stream>>>(
      Ao, wout16t, (int)ROWS, DIM, DIM, 0, out, b_out, nullptr, nullptr, nullptr);
}
